// MultiHeadAttention_11158325035341
// MI455X (gfx1250) — matrix operand order repaired and checked by reading, not compiled, not run on hardware
//
#include <hip/hip_runtime.h>
#include <hip/hip_bf16.h>

// ---------------------------------------------------------------------------
// MHA for B=2, L=2048, D=1024, H=16, DH=64 on gfx1250 (CDNA5, wave32, WMMA).
// Pipeline: f32->f16 converts -> 3 WMMA projection GEMMs (k stored transposed)
// -> flash-attention with fused mask+bias and online softmax -> output GEMM.
// Async global->LDS double-buffered GEMM pipeline (ASYNCcnt), non-temporal
// mask/bias streaming, prefetch of next key tile.
// ---------------------------------------------------------------------------

typedef _Float16 half_t;
typedef __attribute__((ext_vector_type(16))) _Float16 v16h;
typedef __attribute__((ext_vector_type(8)))  _Float16 v8h;
typedef __attribute__((ext_vector_type(4)))  _Float16 v4h;
typedef __attribute__((ext_vector_type(8)))  float    v8f;
typedef __attribute__((ext_vector_type(4)))  float    v4f;
typedef __attribute__((ext_vector_type(4)))  int      v4i;

#define B_  2
#define L_  2048
#define D_  1024
#define H_  16
#define DH_ 64
#define M_  (B_ * L_)   // 4096 rows in all projection GEMMs

// ---- CDNA5 async global->LDS copy support (guarded; falls back to sync) ----
#if defined(__has_builtin)
#if __has_builtin(__builtin_amdgcn_global_load_async_to_lds_b128) && \
    __has_builtin(__builtin_amdgcn_s_wait_asynccnt)
#define HAS_ASYNC_LDS 1
#endif
#endif
#ifndef HAS_ASYNC_LDS
#define HAS_ASYNC_LDS 0
#endif

#if HAS_ASYNC_LDS
typedef __attribute__((address_space(1))) v4i gv4i;   // global int4
typedef __attribute__((address_space(3))) v4i lv4i;   // LDS int4
static __device__ __forceinline__ void async_copy16(const half_t* g, half_t* l) {
  __builtin_amdgcn_global_load_async_to_lds_b128((gv4i*)g, (lv4i*)l, 0, 0);
}
#endif

static __device__ __forceinline__ v16h cat16(v8h lo, v8h hi) {
  return __builtin_shufflevector(lo, hi, 0,1,2,3,4,5,6,7,8,9,10,11,12,13,14,15);
}

// B-fragment: column n = lane&15, K = {0..7,16..23} (lanes 0-15) / {8..15,24..31} (lanes 16-31), strided gather.
static __device__ __forceinline__ v16h load_b16(const half_t* p, int stride, int lane) {
  const half_t* q = p + (lane & 15) + (size_t)((lane >> 4) << 3) * stride;
  v16h r;
#pragma unroll
  for (int s = 0; s < 8; ++s) {
    r[s]     = q[(size_t)s * stride];
    r[s + 8] = q[(size_t)(16 + s) * stride];
  }
  return r;
}

// A-fragment (16x32 f16) per CDNA5 layout:
// lanes 0-15: row = base+lane, K-halves {0..7, 16..23}
// lanes 16-31: row = base+lane-16, K-halves {8..15, 24..31}
static __device__ __forceinline__ v16h load_a_frag(const half_t* lds, int stride,
                                                   int rowBase, int kOff, int lane) {
  const int r  = rowBase + (lane & 15);
  const int kb = kOff + ((lane >> 4) << 3);
  const half_t* p = lds + r * stride + kb;
  v8h lo = *(const v8h*)p;          // K = kb .. kb+7
  v8h hi = *(const v8h*)(p + 16);   // K = kb+16 .. kb+23
  return cat16(lo, hi);
}

static __device__ __forceinline__ v8f wmma_f16(v16h a, v16h b, v8f c) {
  return __builtin_amdgcn_wmma_f32_16x16x32_f16(false, a, false, b,
                                                (short)0, c, false, false);
}

// ---------------------------------------------------------------------------
// f32 -> f16 conversion, 4 elements per thread.
// ---------------------------------------------------------------------------
__global__ void cvt_f32_f16(const float* __restrict__ src,
                            half_t* __restrict__ dst, int n4) {
  int i = blockIdx.x * blockDim.x + threadIdx.x;
  if (i < n4) {
    v4f x = *(const v4f*)(src + (size_t)i * 4);
    v4h y;
    y[0] = (half_t)x[0]; y[1] = (half_t)x[1];
    y[2] = (half_t)x[2]; y[3] = (half_t)x[3];
    *(v4h*)(dst + (size_t)i * 4) = y;
  }
}

// ---------------------------------------------------------------------------
// GEMM: Y = X(4096x1024,f16) @ W(1024x1024,f16) + bias(f32)
// 128x128 block tile, BK=32, 256 threads (8 waves), each wave 32x64.
// Double-buffered async global->LDS pipeline when available.
// MODE 0: f16 out, head-split (B,H,L,DH)            (q, v projections)
// MODE 1: f16 out, head-split transposed (B,H,DH,L) (k projection)
// MODE 2: f32 out, row-major, non-temporal          (final Wo GEMM -> d_out)
// ---------------------------------------------------------------------------
template <int MODE>
__global__ __launch_bounds__(256)
void gemm128(const half_t* __restrict__ X, const half_t* __restrict__ W,
             const float* __restrict__ bias, half_t* __restrict__ dstH,
             float* __restrict__ dstF) {
  __shared__ half_t ldsX[2][128 * 40];   // 128 rows x 32 K-halves, stride 40
  __shared__ half_t ldsW[2][32 * 136];   // 32 K-rows x 128 N-halves, stride 136

  const int tid  = threadIdx.x;
  const int lane = tid & 31;
  const int w    = tid >> 5;
  const int wr   = w & 3;     // wave row group (32 rows each)
  const int wc   = w >> 2;    // wave col group (64 cols each)
  const int m0   = blockIdx.x * 128;
  const int n0   = blockIdx.y * 128;

  v8f acc[2][4];
#pragma unroll
  for (int ti = 0; ti < 2; ++ti)
#pragma unroll
    for (int j = 0; j < 4; ++j) acc[ti][j] = (v8f)0.0f;

  const int xr = tid >> 1, xp = (tid & 1) * 16;   // X tile: 2 thr/row, 16 halves
  const int kr = tid >> 3, wp = (tid & 7) * 16;   // W tile: 8 thr/row, 16 halves
  const half_t* gX = X + (size_t)(m0 + xr) * D_ + xp;       // + kb
  const half_t* gW = W + (size_t)kr * D_ + n0 + wp;         // + kb*D_

#if HAS_ASYNC_LDS
  // prime the pipeline: tile 0 into buffer 0 (4 x b128 async ops per thread)
  async_copy16(gX,     &ldsX[0][xr * 40 + xp]);
  async_copy16(gX + 8, &ldsX[0][xr * 40 + xp + 8]);
  async_copy16(gW,     &ldsW[0][kr * 136 + wp]);
  async_copy16(gW + 8, &ldsW[0][kr * 136 + wp + 8]);
#endif

  int cur = 0;
  for (int kb = 0; kb < D_; kb += 32, cur ^= 1) {
#if HAS_ASYNC_LDS
    if (kb + 32 < D_) {
      const half_t* nx = gX + kb + 32;
      const half_t* nw = gW + (size_t)(kb + 32) * D_;
      async_copy16(nx,     &ldsX[cur ^ 1][xr * 40 + xp]);
      async_copy16(nx + 8, &ldsX[cur ^ 1][xr * 40 + xp + 8]);
      async_copy16(nw,     &ldsW[cur ^ 1][kr * 136 + wp]);
      async_copy16(nw + 8, &ldsW[cur ^ 1][kr * 136 + wp + 8]);
      __builtin_amdgcn_s_wait_asynccnt(4);   // current tile done, next in flight
    } else {
      __builtin_amdgcn_s_wait_asynccnt(0);
    }
#else
    const half_t* gx = gX + kb;
    *(v8h*)&ldsX[cur][xr * 40 + xp]     = *(const v8h*)gx;
    *(v8h*)&ldsX[cur][xr * 40 + xp + 8] = *(const v8h*)(gx + 8);
    const half_t* gw = gW + (size_t)kb * D_;
    *(v8h*)&ldsW[cur][kr * 136 + wp]     = *(const v8h*)gw;
    *(v8h*)&ldsW[cur][kr * 136 + wp + 8] = *(const v8h*)(gw + 8);
#endif
    __syncthreads();

    v16h a0 = load_a_frag(ldsX[cur], 40, wr * 32,      0, lane);
    v16h a1 = load_a_frag(ldsX[cur], 40, wr * 32 + 16, 0, lane);
#pragma unroll
    for (int j = 0; j < 4; ++j) {
      v16h b = load_b16(&ldsW[cur][wc * 64 + j * 16], 136, lane); // lane&15 = N col
      acc[0][j] = wmma_f16(a0, b, acc[0][j]);
      acc[1][j] = wmma_f16(a1, b, acc[1][j]);
    }
    __syncthreads();
  }

  // C/D layout: lane -> N column (lane&15), VGPR v -> row v (+8 for lanes 16-31)
  const int colb = lane & 15;
  const int hs   = (lane >> 4) * 8;
#pragma unroll
  for (int ti = 0; ti < 2; ++ti)
#pragma unroll
    for (int j = 0; j < 4; ++j)
#pragma unroll
      for (int v = 0; v < 8; ++v) {
        const int m = m0 + wr * 32 + ti * 16 + v + hs;
        const int n = n0 + wc * 64 + j * 16 + colb;
        const float val = acc[ti][j][v] + bias[n];
        if (MODE == 2) {
          __builtin_nontemporal_store(val, &dstF[(size_t)m * D_ + n]);
        } else {
          const int bb = m >> 11, l = m & (L_ - 1);
          const int hh = n >> 6,  d = n & (DH_ - 1);
          const size_t idx = (MODE == 0)
              ? (((size_t)(bb * H_ + hh) * L_ + l) * DH_ + d)
              : (((size_t)(bb * H_ + hh) * DH_ + d) * L_ + l);
          dstH[idx] = (half_t)val;
        }
      }
}

// ---------------------------------------------------------------------------
// Flash attention, one (b,h) head slice per blockIdx.y, 64-query tile per
// blockIdx.x, 128 threads (4 waves; wave w owns query rows [16w,16w+16)).
// q: (B,H,L,DH) f16, kT: (B,H,DH,L) f16, v: (B,H,L,DH) f16.
// mask (B,1,L,L) i32 and bias (B,H,L,L) f32 fused at score time (streamed
// non-temporal, next tile prefetched).
// ---------------------------------------------------------------------------
__global__ __launch_bounds__(128)
void attn64(const half_t* __restrict__ qh, const half_t* __restrict__ kth,
            const half_t* __restrict__ vh, const int* __restrict__ mask,
            const float* __restrict__ bias, half_t* __restrict__ outh) {
  __shared__ half_t ldsQ[64 * 72];
  __shared__ float  ldsS[64 * 68];
  __shared__ half_t ldsP[64 * 72];
  __shared__ float  ldsM[64];
  __shared__ float  ldsL[64];
  __shared__ float  ldsA[64];

  const int tid  = threadIdx.x;
  const int lane = tid & 31;
  const int w    = tid >> 5;
  const int bh   = blockIdx.y;
  const int b    = bh >> 4;          // bh / H
  const int q0   = blockIdx.x * 64;
  const int colb = lane & 15;
  const int hs   = (lane >> 4) * 8;

  { // stage the 64x64 f16 q tile into LDS (stride 72)
    const int r = tid >> 1, part = (tid & 1) * 32;
    const half_t* g = qh + ((size_t)bh * L_ + q0 + r) * DH_ + part;
    half_t* l = &ldsQ[r * 72 + part];
#if HAS_ASYNC_LDS
#pragma unroll
    for (int c = 0; c < 4; ++c) async_copy16(g + c * 8, l + c * 8);
    __builtin_amdgcn_s_wait_asynccnt(0);
#else
#pragma unroll
    for (int c = 0; c < 4; ++c) *(v8h*)(l + c * 8) = *(const v8h*)(g + c * 8);
#endif
  }
  if (tid < 64) { ldsM[tid] = -1e30f; ldsL[tid] = 0.0f; }
  __syncthreads();

  v8f o[4];
#pragma unroll
  for (int j = 0; j < 4; ++j) o[j] = (v8f)0.0f;
  const float scale = 0.125f;   // 1/sqrt(DH)

  for (int kt = 0; kt < L_ / 64; ++kt) {
    const int key0 = kt * 64;

    // prefetch next tile's bias/mask (one 128B line per thread each)
    if (kt + 1 < L_ / 64) {
      const int pr = tid >> 1, pc = (tid & 1) * 32;
      __builtin_prefetch(&bias[((size_t)bh * L_ + q0 + pr) * L_ + key0 + 64 + pc], 0, 0);
      __builtin_prefetch(&mask[((size_t)b  * L_ + q0 + pr) * L_ + key0 + 64 + pc], 0, 0);
    }

    // ---- S = q @ kT (16x64 strip per wave), K contraction over DH=64 ----
    v8f sa[4];
#pragma unroll
    for (int j = 0; j < 4; ++j) sa[j] = (v8f)0.0f;
#pragma unroll
    for (int s = 0; s < 2; ++s) {
      v16h a = load_a_frag(ldsQ, 72, w * 16, s * 32, lane);
      const half_t* kbase = kth + ((size_t)bh * DH_ + s * 32) * L_ + key0;
#pragma unroll
      for (int j = 0; j < 4; ++j) {
        v16h bf = load_b16(kbase + j * 16, L_, lane);   // lane&15 = key col
        sa[j] = wmma_f16(a, bf, sa[j]);
      }
    }

    // ---- scale + mask + bias (streamed NT), spill scores to LDS ----
#pragma unroll
    for (int j = 0; j < 4; ++j)
#pragma unroll
      for (int v = 0; v < 8; ++v) {
        const int rq  = w * 16 + v + hs;
        const int gq  = q0 + rq;
        const int key = key0 + j * 16 + colb;
        float sv = sa[j][v] * scale;
        const int mk = __builtin_nontemporal_load(&mask[((size_t)b * L_ + gq) * L_ + key]);
        sv = (mk == 0) ? -10000.0f : sv;
        sv += __builtin_nontemporal_load(&bias[((size_t)bh * L_ + gq) * L_ + key]);
        ldsS[rq * 68 + j * 16 + colb] = sv;
      }
    __syncthreads();

    // ---- online softmax, one thread per query row ----
    if (tid < 64) {
      const float* srow = &ldsS[tid * 68];
      float tm = srow[0];
#pragma unroll 4
      for (int i = 1; i < 64; ++i) tm = fmaxf(tm, srow[i]);
      const float mold = ldsM[tid];
      const float mnew = fmaxf(mold, tm);
      const float alpha = __expf(mold - mnew);
      float ssum = 0.0f;
      half_t* prow = &ldsP[tid * 72];
#pragma unroll 4
      for (int i = 0; i < 64; ++i) {
        const float e = __expf(srow[i] - mnew);
        ssum += e;
        prow[i] = (half_t)e;
      }
      ldsL[tid] = ldsL[tid] * alpha + ssum;
      ldsM[tid] = mnew;
      ldsA[tid] = alpha;
    }
    __syncthreads();

    // ---- rescale running O, then O += P @ V ----
#pragma unroll
    for (int j = 0; j < 4; ++j)
#pragma unroll
      for (int v = 0; v < 8; ++v) o[j][v] *= ldsA[w * 16 + v + hs];
#pragma unroll
    for (int s = 0; s < 2; ++s) {
      v16h a = load_a_frag(ldsP, 72, w * 16, s * 32, lane);
      const half_t* vbase = vh + ((size_t)bh * L_ + key0 + s * 32) * DH_;
#pragma unroll
      for (int j = 0; j < 4; ++j) {
        v16h bf = load_b16(vbase + j * 16, DH_, lane);
        o[j] = wmma_f16(a, bf, o[j]);
      }
    }
    __syncthreads();
  }

  // ---- normalize and write attention output in (B, L, D) f16 ----
#pragma unroll
  for (int j = 0; j < 4; ++j)
#pragma unroll
    for (int v = 0; v < 8; ++v) {
      const int rq = w * 16 + v + hs;
      const int gq = q0 + rq;
      const float ov = o[j][v] / ldsL[rq];
      const int n = (bh & 15) * DH_ + j * 16 + colb;
      outh[((size_t)b * L_ + gq) * D_ + n] = (half_t)ov;
    }
}

// ---------------------------------------------------------------------------
extern "C" void kernel_launch(void* const* d_in, const int* in_sizes, int n_in,
                              void* d_out, int out_size, void* d_ws, size_t ws_size,
                              hipStream_t stream) {
  const float* Q    = (const float*)d_in[0];
  const float* K    = (const float*)d_in[1];
  const float* V    = (const float*)d_in[2];
  const int*   mask = (const int*)d_in[3];
  const float* bias = (const float*)d_in[4];
  const float* Wq   = (const float*)d_in[5];
  const float* bq   = (const float*)d_in[6];
  const float* Wk   = (const float*)d_in[7];
  const float* bk   = (const float*)d_in[8];
  const float* Wv   = (const float*)d_in[9];
  const float* bv   = (const float*)d_in[10];
  const float* Wo   = (const float*)d_in[11];
  const float* bo   = (const float*)d_in[12];
  float* out = (float*)d_out;

  // Workspace carve (64 MB total, all f16)
  char* ws = (char*)d_ws;
  const size_t SZ_XD = (size_t)M_ * D_ * sizeof(half_t);   // 8 MB
  const size_t SZ_W  = (size_t)D_ * D_ * sizeof(half_t);   // 2 MB
  half_t* Qh      = (half_t*)ws;  ws += SZ_XD;
  half_t* Kh      = (half_t*)ws;  ws += SZ_XD;
  half_t* Vh      = (half_t*)ws;  ws += SZ_XD;
  half_t* Wqh     = (half_t*)ws;  ws += SZ_W;
  half_t* Wkh     = (half_t*)ws;  ws += SZ_W;
  half_t* Wvh     = (half_t*)ws;  ws += SZ_W;
  half_t* Woh     = (half_t*)ws;  ws += SZ_W;
  half_t* qproj   = (half_t*)ws;  ws += SZ_XD;   // (B,H,L,DH)
  half_t* ktproj  = (half_t*)ws;  ws += SZ_XD;   // (B,H,DH,L)
  half_t* vproj   = (half_t*)ws;  ws += SZ_XD;   // (B,H,L,DH)
  half_t* attnout = (half_t*)ws;  ws += SZ_XD;   // (B,L,D)

  // f32 -> f16 conversions
  const int n4x = M_ * D_ / 4;   // 1,048,576
  const int n4w = D_ * D_ / 4;   //   262,144
  cvt_f32_f16<<<n4x / 256, 256, 0, stream>>>(Q,  Qh,  n4x);
  cvt_f32_f16<<<n4x / 256, 256, 0, stream>>>(K,  Kh,  n4x);
  cvt_f32_f16<<<n4x / 256, 256, 0, stream>>>(V,  Vh,  n4x);
  cvt_f32_f16<<<n4w / 256, 256, 0, stream>>>(Wq, Wqh, n4w);
  cvt_f32_f16<<<n4w / 256, 256, 0, stream>>>(Wk, Wkh, n4w);
  cvt_f32_f16<<<n4w / 256, 256, 0, stream>>>(Wv, Wvh, n4w);
  cvt_f32_f16<<<n4w / 256, 256, 0, stream>>>(Wo, Woh, n4w);

  // Projections
  dim3 gg(M_ / 128, D_ / 128);   // (32, 8)
  gemm128<0><<<gg, 256, 0, stream>>>(Qh, Wqh, bq, qproj,  nullptr);
  gemm128<1><<<gg, 256, 0, stream>>>(Kh, Wkh, bk, ktproj, nullptr);
  gemm128<0><<<gg, 256, 0, stream>>>(Vh, Wvh, bv, vproj,  nullptr);

  // Flash attention
  dim3 ga(L_ / 64, B_ * H_);     // (32, 32)
  attn64<<<ga, 128, 0, stream>>>(qproj, ktproj, vproj, mask, bias, attnout);

  // Output projection -> f32 d_out
  gemm128<2><<<gg, 256, 0, stream>>>(attnout, Woh, bo, nullptr, out);
}
